// CublasTransformerBlock_2250562863806
// MI455X (gfx1250) — matrix operand order repaired and checked by reading, not compiled, not run on hardware
//
#include <hip/hip_runtime.h>
#include <hip/hip_bf16.h>
#include <math.h>

typedef __bf16 bf16;
typedef __attribute__((ext_vector_type(16))) __bf16 v16bf;
typedef __attribute__((ext_vector_type(8)))  float   v8f;
typedef __attribute__((ext_vector_type(4)))  int     v4i;

#define TB_DIM   1024
#define TB_MLP   4096
#define TB_SEQ   2048
#define TB_BATCH 4
#define TB_ROWS  (TB_BATCH * TB_SEQ)   // 8192

// ---------------------------------------------------------------------------
// CDNA5 async global->LDS path (confirmed working on this toolchain)
// ---------------------------------------------------------------------------
#if defined(__AMDGCN__) && __has_builtin(__builtin_amdgcn_global_load_async_to_lds_b128)
#define GEMM_ASYNC 1
#else
#define GEMM_ASYNC 0
#endif

#if GEMM_ASYNC
typedef __attribute__((address_space(1))) v4i as1_v4i;  // HIP: __device__ AS
typedef __attribute__((address_space(3))) v4i as3_v4i;  // HIP: __shared__ AS
__device__ __forceinline__ void async_copy16(const void* g, void* l) {
  __builtin_amdgcn_global_load_async_to_lds_b128(
      (as1_v4i*)(unsigned long long)g,
      (as3_v4i*)(unsigned int)(unsigned long long)l, 0, 0);
}
__device__ __forceinline__ void wait_async0() {
#if __has_builtin(__builtin_amdgcn_s_wait_asynccnt)
  __builtin_amdgcn_s_wait_asynccnt(0);
#else
  asm volatile("s_wait_asynccnt 0x0" ::: "memory");
#endif
}
__device__ __forceinline__ void wait_async8() {   // tile t done, tile t+1 in flight
#if __has_builtin(__builtin_amdgcn_s_wait_asynccnt)
  __builtin_amdgcn_s_wait_asynccnt(8);
#else
  asm volatile("s_wait_asynccnt 0x8" ::: "memory");
#endif
}
#endif

// ---------------------------------------------------------------------------
// Generic batched GEMM:  C[M,N] = act(alpha * A[M,K] @ Bt[N,K]^T + bias [+res])
// A row-major (lda=K), B supplied PRE-TRANSPOSED as Bt[N][K] (ldb=K).
// OM: 0 = f32 row-major, 1 = bf16 row-major, 2 = bf16 transposed (n*ldct+m)
// ---------------------------------------------------------------------------
struct GemmP {
  const bf16* A;
  const bf16* B;            // Bt: [N][K]
  void*       C;
  const float* bias;        // [N]
  const float* residual;    // row-major like C
  int M, N, K;
  long long sA, sB, sC, sR; // per-batch strides (elements)
  float alpha;
  int ldct;
};

#define BM 128
#define BN 128
#define BK 32
#define LDP 40       // padded LDS pitch in bf16 elements (80B -> conflict-free)
#define BUFE (BM * LDP)

template <int OM, bool BIAS, bool RES, bool GELU>
__global__ __launch_bounds__(128) void gemm_bf16_wmma(GemmP p) {
  __shared__ bf16 As[2 * BUFE];   // double-buffered tiles
  __shared__ bf16 Bs[2 * BUFE];

  const int tid  = threadIdx.x;       // 128 threads = 4 waves
  const int lane = tid & 31;
  const int wave = tid >> 5;
  const int wm = (wave & 1) * 64;     // wave M offset inside 128x128 tile
  const int wn = (wave >> 1) * 64;    // wave N offset
  const int z  = blockIdx.z;

  const int bm0 = blockIdx.y * BM;
  const int bn0 = blockIdx.x * BN;

  const bf16* Arow = p.A + (long long)z * p.sA + (size_t)(bm0 + tid) * p.K;
  const bf16* Brow = p.B + (long long)z * p.sB + (size_t)(bn0 + tid) * p.K;

  v8f acc[4][4] = {};
  const int nk = p.K / BK;

#if GEMM_ASYNC
  auto issue = [&](int t, int buf) {   // 8 x b128 async copies per thread
    const int k0 = t * BK;
    #pragma unroll
    for (int c = 0; c < 4; ++c) {
      async_copy16(Arow + k0 + c * 8, &As[buf * BUFE + tid * LDP + c * 8]);
      async_copy16(Brow + k0 + c * 8, &Bs[buf * BUFE + tid * LDP + c * 8]);
    }
  };
  issue(0, 0);
#endif

  for (int t = 0; t < nk; ++t) {
    const int c = t & 1;
#if GEMM_ASYNC
    if (t + 1 < nk) {
      issue(t + 1, 1 - c);               // overlap next tile with this compute
      if (t + 2 < nk) {                  // warm GL2 two tiles ahead
        __builtin_prefetch(Arow + (t + 2) * BK, 0, 1);
        __builtin_prefetch(Brow + (t + 2) * BK, 0, 1);
      }
      wait_async8();                     // in-order: first 8 (tile t) landed
    } else {
      wait_async0();
    }
    __syncthreads();                     // tile t visible to all waves
#else
    const int k0 = t * BK;
    uint4 a[4], b[4];
    #pragma unroll
    for (int q = 0; q < 4; ++q) {
      a[q] = *(const uint4*)(Arow + k0 + q * 8);
      b[q] = *(const uint4*)(Brow + k0 + q * 8);
    }
    __syncthreads();
    #pragma unroll
    for (int q = 0; q < 4; ++q) {
      *(uint4*)&As[c * BUFE + tid * LDP + q * 8] = a[q];
      *(uint4*)&Bs[c * BUFE + tid * LDP + q * 8] = b[q];
    }
    __syncthreads();
#endif

    // ---- fragments per the CDNA5 16-bit WMMA layouts ----
    const bf16* AsC = &As[c * BUFE];
    const bf16* BsC = &Bs[c * BUFE];
    union Frag { uint4 u[2]; v16bf v; };
    Frag af[4], bfg[4];
    const int r16 = lane & 15;
    const int akb = (lane < 16) ? 0 : 8;   // A: K {kb..kb+7, kb+16..kb+23}
    const int bkc = (lane < 16) ? 0 : 8;   // B: K {kb..kb+7, kb+16..kb+23}
    #pragma unroll
    for (int q = 0; q < 4; ++q) {
      const int r = wm + q * 16 + r16;
      af[q].u[0] = *(const uint4*)&AsC[r * LDP + akb];
      af[q].u[1] = *(const uint4*)&AsC[r * LDP + akb + 16];
      const int n = wn + q * 16 + r16;
      bfg[q].u[0] = *(const uint4*)&BsC[n * LDP + bkc];
      bfg[q].u[1] = *(const uint4*)&BsC[n * LDP + bkc + 16];
    }

    #pragma unroll
    for (int mt = 0; mt < 4; ++mt)
      #pragma unroll
      for (int nt = 0; nt < 4; ++nt)
        acc[mt][nt] = __builtin_amdgcn_wmma_f32_16x16x32_bf16(
            false, af[mt].v, false, bfg[nt].v, (short)0, acc[mt][nt],
            false, false);

    __syncthreads();   // all waves done reading buf c before it is rewritten
  }

  // ---- epilogue: acc VGPR i -> (M = i + (lane<16?0:8), N = lane&15) ----
  const int rh = (lane < 16) ? 0 : 8;
  const int nc = lane & 15;
  const long long cofs = (long long)z * p.sC;
  const long long rofs = (long long)z * p.sR;
  #pragma unroll
  for (int mt = 0; mt < 4; ++mt) {
    #pragma unroll
    for (int nt = 0; nt < 4; ++nt) {
      const int mb = bm0 + wm + mt * 16 + rh;
      const int n  = bn0 + wn + nt * 16 + nc;
      float bi = 0.f;
      if (BIAS) bi = p.bias[n];
      #pragma unroll
      for (int i = 0; i < 8; ++i) {
        const int m = mb + i;
        float v = acc[mt][nt][i] * p.alpha;
        if (BIAS) v += bi;
        if (RES)  v += p.residual[rofs + (size_t)m * p.N + n];
        if (GELU) v = 0.5f * v * (1.0f + erff(v * 0.70710678118654752f));
        if (OM == 2)
          ((bf16*)p.C)[cofs + (size_t)n * p.ldct + m] = (bf16)v;
        else if (OM == 1)
          ((bf16*)p.C)[cofs + (size_t)m * p.N + n] = (bf16)v;
        else
          ((float*)p.C)[cofs + (size_t)m * p.N + n] = v;
      }
    }
  }
}

// ---------------------------------------------------------------------------
// W[K][N] fp32  ->  Wt[N][K] bf16
// ---------------------------------------------------------------------------
__global__ void transpose_to_bf16(const float* __restrict__ in,
                                  bf16* __restrict__ out, int K, int N) {
  long long idx = (long long)blockIdx.x * 256 + threadIdx.x;
  if (idx >= (long long)K * N) return;
  const int k = (int)(idx / N);
  const int n = (int)(idx % N);
  out[(long long)n * K + k] = (bf16)in[idx];
}

// ---------------------------------------------------------------------------
// Row LayerNorm (biased var, eps=1e-5) -> bf16
// ---------------------------------------------------------------------------
__global__ __launch_bounds__(256) void layernorm_bf16(
    const float* __restrict__ x, const float* __restrict__ gamma,
    const float* __restrict__ beta, bf16* __restrict__ out, int dim) {
  __shared__ float red[256];
  const int row = blockIdx.x, tid = threadIdx.x;
  const float* xr = x + (long long)row * dim;
  float s = 0.f;
  for (int j = tid; j < dim; j += 256) s += xr[j];
  red[tid] = s; __syncthreads();
  for (int o = 128; o > 0; o >>= 1) { if (tid < o) red[tid] += red[tid + o]; __syncthreads(); }
  const float mean = red[0] / dim;
  __syncthreads();
  float vs = 0.f;
  for (int j = tid; j < dim; j += 256) { const float d = xr[j] - mean; vs += d * d; }
  red[tid] = vs; __syncthreads();
  for (int o = 128; o > 0; o >>= 1) { if (tid < o) red[tid] += red[tid + o]; __syncthreads(); }
  const float rstd = rsqrtf(red[0] / dim + 1e-5f);
  bf16* orow = out + (long long)row * dim;
  for (int j = tid; j < dim; j += 256)
    orow[j] = (bf16)(gamma[j] * (xr[j] - mean) * rstd + beta[j]);
}

// ---------------------------------------------------------------------------
// In-place row softmax over bf16 (scale already applied upstream)
// ---------------------------------------------------------------------------
__global__ __launch_bounds__(256) void softmax_bf16_inplace(bf16* a, int cols) {
  __shared__ float red[256];
  const int row = blockIdx.x, tid = threadIdx.x;
  bf16* ar = a + (long long)row * cols;
  float m = -3.0e38f;
  for (int j = tid; j < cols; j += 256) m = fmaxf(m, (float)ar[j]);
  red[tid] = m; __syncthreads();
  for (int o = 128; o > 0; o >>= 1) { if (tid < o) red[tid] = fmaxf(red[tid], red[tid + o]); __syncthreads(); }
  const float rowmax = red[0];
  __syncthreads();
  float s = 0.f;
  for (int j = tid; j < cols; j += 256) s += expf((float)ar[j] - rowmax);
  red[tid] = s; __syncthreads();
  for (int o = 128; o > 0; o >>= 1) { if (tid < o) red[tid] += red[tid + o]; __syncthreads(); }
  const float inv = 1.f / red[0];
  for (int j = tid; j < cols; j += 256)
    ar[j] = (bf16)(expf((float)ar[j] - rowmax) * inv);
}

// ---------------------------------------------------------------------------
template <int OM, bool BIAS, bool RES, bool GELU>
static inline void run_gemm(hipStream_t st, const bf16* A, const bf16* B, void* C,
                            const float* bias, const float* res,
                            int M, int N, int K, int nb,
                            long long sA, long long sB, long long sC, long long sR,
                            float alpha, int ldct) {
  GemmP p;
  p.A = A; p.B = B; p.C = C; p.bias = bias; p.residual = res;
  p.M = M; p.N = N; p.K = K;
  p.sA = sA; p.sB = sB; p.sC = sC; p.sR = sR;
  p.alpha = alpha; p.ldct = ldct;
  dim3 grid(N / BN, M / BM, nb);
  gemm_bf16_wmma<OM, BIAS, RES, GELU><<<grid, 128, 0, st>>>(p);
}

extern "C" void kernel_launch(void* const* d_in, const int* in_sizes, int n_in,
                              void* d_out, int out_size, void* d_ws, size_t ws_size,
                              hipStream_t stream) {
  (void)in_sizes; (void)n_in; (void)out_size; (void)ws_size;

  const float* x   = (const float*)d_in[0];
  const float* g1  = (const float*)d_in[1];
  const float* be1 = (const float*)d_in[2];
  const float* g2  = (const float*)d_in[3];
  const float* be2 = (const float*)d_in[4];
  const float* Wq  = (const float*)d_in[5];
  const float* bq  = (const float*)d_in[6];
  const float* Wk  = (const float*)d_in[7];
  const float* bk  = (const float*)d_in[8];
  const float* Wv  = (const float*)d_in[9];
  const float* bv  = (const float*)d_in[10];
  const float* W1  = (const float*)d_in[11];
  const float* b1  = (const float*)d_in[12];
  const float* W2  = (const float*)d_in[13];
  const float* b2  = (const float*)d_in[14];
  float* out = (float*)d_out;

  // ---- workspace carve-up (256B aligned) ----
  char* wsb = (char*)d_ws;
  size_t off = 0;
  auto take = [&](size_t bytes) -> char* {
    char* p = wsb + off;
    off += (bytes + 255) & ~(size_t)255;
    return p;
  };
  bf16*  hbf  = (bf16*)take((size_t)TB_ROWS * TB_DIM * 2);           // LN1/LN2 out
  bf16*  wqt  = (bf16*)take((size_t)TB_DIM * TB_DIM * 2);
  bf16*  wkt  = (bf16*)take((size_t)TB_DIM * TB_DIM * 2);
  bf16*  wvt  = (bf16*)take((size_t)TB_DIM * TB_DIM * 2);
  bf16*  w1t  = (bf16*)take((size_t)TB_DIM * TB_MLP * 2);            // [4096][1024]
  bf16*  w2t  = (bf16*)take((size_t)TB_MLP * TB_DIM * 2);            // [1024][4096]
  bf16*  Qb   = (bf16*)take((size_t)TB_ROWS * TB_DIM * 2);
  bf16*  Kb   = (bf16*)take((size_t)TB_ROWS * TB_DIM * 2);
  bf16*  Vt   = (bf16*)take((size_t)TB_BATCH * TB_DIM * TB_SEQ * 2); // [b][d][t]
  bf16*  attn = (bf16*)take((size_t)TB_BATCH * TB_SEQ * TB_SEQ * 2); // scores->attn
  float* x2   = (float*)take((size_t)TB_ROWS * TB_DIM * 4);          // attn out
  bf16*  gmid = (bf16*)take((size_t)TB_ROWS * TB_MLP * 2);           // gelu(h@W1+b1)

  const long long SD = (long long)TB_SEQ * TB_DIM;
  const long long SS = (long long)TB_SEQ * TB_SEQ;
  const long long DS = (long long)TB_DIM * TB_SEQ;

  // ---- 1. pre-transpose + convert weights to bf16 ----
  {
    const int nsq = (TB_DIM * TB_DIM + 255) / 256;
    transpose_to_bf16<<<nsq, 256, 0, stream>>>(Wq, wqt, TB_DIM, TB_DIM);
    transpose_to_bf16<<<nsq, 256, 0, stream>>>(Wk, wkt, TB_DIM, TB_DIM);
    transpose_to_bf16<<<nsq, 256, 0, stream>>>(Wv, wvt, TB_DIM, TB_DIM);
    const int nmlp = (TB_DIM * TB_MLP + 255) / 256;
    transpose_to_bf16<<<nmlp, 256, 0, stream>>>(W1, w1t, TB_DIM, TB_MLP);
    transpose_to_bf16<<<nmlp, 256, 0, stream>>>(W2, w2t, TB_MLP, TB_DIM);
  }

  // ---- 2. h1 = LN(x) ----
  layernorm_bf16<<<TB_ROWS, 256, 0, stream>>>(x, g1, be1, hbf, TB_DIM);

  // ---- 3-5. Q, K row-major; V stored transposed per batch ----
  run_gemm<1, true, false, false>(stream, hbf, wqt, Qb, bq, nullptr,
      TB_ROWS, TB_DIM, TB_DIM, 1, 0, 0, 0, 0, 1.0f, 0);
  run_gemm<1, true, false, false>(stream, hbf, wkt, Kb, bk, nullptr,
      TB_ROWS, TB_DIM, TB_DIM, 1, 0, 0, 0, 0, 1.0f, 0);
  run_gemm<2, true, false, false>(stream, hbf, wvt, Vt, bv, nullptr,
      TB_SEQ, TB_DIM, TB_DIM, TB_BATCH, SD, 0, DS, 0, 1.0f, TB_SEQ);

  // ---- 6. scores = (Q @ K^T) / 32  (K row-major *is* Bt here) ----
  run_gemm<1, false, false, false>(stream, Qb, Kb, attn, nullptr, nullptr,
      TB_SEQ, TB_SEQ, TB_DIM, TB_BATCH, SD, SD, SS, 0, 1.0f / 32.0f, 0);

  // ---- 7. softmax in place ----
  softmax_bf16_inplace<<<TB_ROWS, 256, 0, stream>>>(attn, TB_SEQ);

  // ---- 8. x2 = attn @ V + x  (Vt[b][d][t] is Bt) ----
  run_gemm<0, false, true, false>(stream, attn, Vt, x2, nullptr, x,
      TB_SEQ, TB_DIM, TB_SEQ, TB_BATCH, SS, DS, SD, SD, 1.0f, 0);

  // ---- 9. h2 = LN(x2) ----
  layernorm_bf16<<<TB_ROWS, 256, 0, stream>>>(x2, g2, be2, hbf, TB_DIM);

  // ---- 10. gmid = gelu(h2 @ W1 + b1) ----
  run_gemm<1, true, false, true>(stream, hbf, w1t, gmid, b1, nullptr,
      TB_ROWS, TB_MLP, TB_DIM, 1, 0, 0, 0, 0, 1.0f, 0);

  // ---- 11. out = gmid @ W2 + b2 + x2 ----
  run_gemm<0, true, true, false>(stream, gmid, w2t, out, b2, x2,
      TB_ROWS, TB_DIM, TB_MLP, 1, 0, 0, 0, 0, 1.0f, 0);
}
